// FROGNetSHG_43078521979572
// MI455X (gfx1250) — matrix operand order repaired and checked by reading, not compiled, not run on hardware
//
#include <hip/hip_runtime.h>
#include <math.h>

// CDNA5 / gfx1250: wave32, WMMA 16x16x32 f16 -> f32 accumulate.
typedef __attribute__((ext_vector_type(16))) _Float16 v16h;
typedef __attribute__((ext_vector_type(8)))  float    v8f;

union V16 { v16h v; _Float16 e[16]; };
union V8  { v8f  v; float    e[8];  };

#define NPTS 256

__global__ __launch_bounds__(256)
void frog_shg_kernel(const float* __restrict__ x, float* __restrict__ out)
{
    __shared__ float2   Esh[NPTS];    // normalized complex field (2 KB)
    __shared__ unsigned Wt[NPTS];     // packed half2 twiddles W[m]=exp(-2*pi*i*m/256) (1 KB)
    __shared__ float    red[NPTS];    // reduction scratch (1 KB)
    // A operand (gate) in final WMMA A-register image, per wave: [wave][chunk][lane][vgpr d]
    // packed half2 (elements 2d, 2d+1). 64 KB each plane.
    __shared__ __attribute__((aligned(32))) unsigned AtR[8][8][32][8];
    __shared__ __attribute__((aligned(32))) unsigned AtI[8][8][32][8];
    // B operand (DFT matrix) for current tk, shared by all waves: [chunk][lane][vgpr d]. 8 KB each.
    __shared__ __attribute__((aligned(32))) unsigned BtR[8][32][8];
    __shared__ __attribute__((aligned(32))) unsigned BtI[8][32][8];

    const int t = threadIdx.x;
    const int b = blockIdx.x;
    const size_t base = (size_t)b * (NPTS * NPTS);

    // ---- Phase 1: load E, normalize by max |E|, build twiddle table ----
    float re = x[(size_t)b * (2 * NPTS) + t];
    float im = x[(size_t)b * (2 * NPTS) + NPTS + t];
    red[t] = re * re + im * im;
    __syncthreads();
    for (int s = 128; s > 0; s >>= 1) {
        if (t < s) red[t] = fmaxf(red[t], red[t + s]);
        __syncthreads();
    }
    float maxabs = fmaxf(sqrtf(red[0]), 1e-20f);
    float rinv = 1.0f / maxabs;
    Esh[t] = make_float2(re * rinv, im * rinv);

    // W[m] = cos(th) + i*sin(th), th = -2*pi*m/256  (packed: lo=re, hi=im)
    float ang = -6.283185307179586f * (float)t * (1.0f / 256.0f);
    float sn, cs;
    sincosf(ang, &sn, &cs);
    union { unsigned u; _Float16 h[2]; } pk;
    pk.h[0] = (_Float16)cs;
    pk.h[1] = (_Float16)sn;
    Wt[t] = pk.u;
    __syncthreads();

    // ---- Phase 2: spec = gate @ W via WMMA ----
    const int wave = t >> 5;
    const int lane = t & 31;
    const int mrow = lane & 15;       // A row / C column within tile
    const int hi   = lane >> 4;       // lane-group select

    float lmax = 0.0f;

    // Each wave owns delay-row blocks ti = 2*wave, 2*wave+1.
    for (int tt = 0; tt < 2; ++tt) {
        const int ti = wave * 2 + tt;
        const int i0 = ti * 16;
        const int irow = i0 + mrow;   // this lane's A row (delay index)

        // Build A image in LDS once per row-block (each wave touches only its slice;
        // same-wave LDS ops stay in order, so no barrier needed).
        // A tile 16x32 f16 layout (ISA 7.12.2):
        //   lanes 0-15: K in {0..7,16..23}; lanes 16-31: K in {8..15,24..31}
        #pragma unroll 1
        for (int chunk = 0; chunk < 8; ++chunk) {
            const int j0 = chunk * 32;
            V16 ar, ai;
            #pragma unroll
            for (int e2 = 0; e2 < 16; ++e2) {
                int kloc = e2 + (e2 & 8) + (hi << 3);
                int j = j0 + kloc;
                float2 E1 = Esh[j];
                float2 E2 = Esh[(j - irow) & 255];
                float gr = E1.x * E2.x - E1.y * E2.y;
                float gi = E1.x * E2.y + E1.y * E2.x;
                ar.e[e2] = (_Float16)gr;
                ai.e[e2] = (_Float16)gi;
            }
            *(v16h*)&AtR[wave][chunk][lane][0] = ar.v;
            *(v16h*)&AtI[wave][chunk][lane][0] = ai.v;
        }

        for (int tk = 0; tk < 16; ++tk) {
            // -- Cooperative B-tile build for this tk (identical for all waves) --
            __syncthreads();   // previous consumers done before overwrite
            {
                const int bcol = tk * 16 + mrow;
                unsigned vR[8], vI[8];
                #pragma unroll
                for (int d = 0; d < 8; ++d) {
                    int j = wave * 32 + 2 * d + ((2 * d) & 8) + (hi << 3);
                    unsigned w1 = Wt[(j * bcol) & 255];
                    unsigned w2 = Wt[((j + 1) * bcol) & 255];
                    vR[d] = (w1 & 0xFFFFu) | (w2 << 16);
                    vI[d] = (w1 >> 16) | (w2 & 0xFFFF0000u);
                }
                #pragma unroll
                for (int d = 0; d < 8; ++d) {
                    BtR[wave][lane][d] = vR[d];
                    BtI[wave][lane][d] = vI[d];
                }
            }
            __syncthreads();   // B tiles ready

            const int kcol = tk * 16 + mrow;   // this lane's C column (freq index)
            V8 P = {}, Q = {}, R = {}, S = {};  // ArBr, AiBi, ArBi, AiBr

            #pragma unroll 2   // limit load prefetch depth -> keep VGPRs < 256
            for (int chunk = 0; chunk < 8; ++chunk) {
                v16h arv = *(const v16h*)&AtR[wave][chunk][lane][0];
                v16h aiv = *(const v16h*)&AtI[wave][chunk][lane][0];
                v16h brv = *(const v16h*)&BtR[chunk][lane][0];
                v16h biv = *(const v16h*)&BtI[chunk][lane][0];
                P.v = __builtin_amdgcn_wmma_f32_16x16x32_f16(false, arv, false, brv, (short)0, P.v, false, false);
                Q.v = __builtin_amdgcn_wmma_f32_16x16x32_f16(false, aiv, false, biv, (short)0, Q.v, false, false);
                R.v = __builtin_amdgcn_wmma_f32_16x16x32_f16(false, arv, false, biv, (short)0, R.v, false, false);
                S.v = __builtin_amdgcn_wmma_f32_16x16x32_f16(false, aiv, false, brv, (short)0, S.v, false, false);
            }

            // Intensity + fused fftshift(-1,-2) and reverse(-1), unnormalized store
            #pragma unroll
            for (int vv = 0; vv < 8; ++vv) {
                float sre = P.e[vv] - Q.e[vv];
                float sim = R.e[vv] + S.e[vv];
                float I = sre * sre + sim * sim;
                int i = i0 + vv + (hi << 3);           // C/D layout: M = v + 8*hi
                int orow = (i + 128) & 255;            // fftshift rows
                int ocol = (127 - kcol) & 255;         // fftshift + reverse cols
                out[base + (size_t)orow * NPTS + ocol] = I;
                lmax = fmaxf(lmax, I);
            }
        }
    }

    // ---- Phase 3: per-batch max + in-place normalize ----
    __syncthreads();
    red[t] = lmax;
    __syncthreads();
    for (int s = 128; s > 0; s >>= 1) {
        if (t < s) red[t] = fmaxf(red[t], red[t + s]);
        __syncthreads();
    }
    float sc = 1.0f / fmaxf(red[0], 1e-20f);
    __threadfence();          // make phase-2 global stores visible block-wide
    __syncthreads();
    for (int q = 0; q < 256; ++q) {
        size_t pos = base + (size_t)q * NPTS + t;  // coalesced across threads
        out[pos] = out[pos] * sc;
    }
}

extern "C" void kernel_launch(void* const* d_in, const int* in_sizes, int n_in,
                              void* d_out, int out_size, void* d_ws, size_t ws_size,
                              hipStream_t stream) {
    (void)n_in; (void)out_size; (void)d_ws; (void)ws_size;
    const float* x = (const float*)d_in[0];
    float* out = (float*)d_out;
    int B = in_sizes[0] / (2 * NPTS);   // 512 for the reference shapes
    frog_shg_kernel<<<B, 256, 0, stream>>>(x, out);
}
